// AttentionRouteModel_90220083019999
// MI455X (gfx1250) — compile-verified, matrix operand order checked by reading, not run on hardware
//
#include <hip/hip_runtime.h>
#include <hip/hip_bf16.h>

// ---------------------------------------------------------------------------
// AttentionRouteModel for MI455X (gfx1250, wave32, WMMA).
// - Encoder GEMMs: LDS-staged A strip (bf16), per-wave 16x16 tiles via
//   v_wmma_f32_16x16x32_bf16, K templated so the K-loop unrolls (4 wmma/body).
// - Attention: flash-style per-(b,head) block, K/V in LDS bf16, QK^T and PV
//   via WMMA, online softmax with wave32 shuffle reductions.
// - Decode: 64 independent blocks running the 512 sequential pointer steps.
// ---------------------------------------------------------------------------

typedef __attribute__((ext_vector_type(16))) __bf16 v16bf;
typedef __attribute__((ext_vector_type(8)))  __bf16 v8bf;
typedef __attribute__((ext_vector_type(8)))  float  v8f;

constexpr int B_  = 64;
constexpr int N_  = 512;
constexpr int S_  = 513;          // sequence length N+1
constexpr int D_  = 128;
constexpr int H_  = 8;
constexpr int DH_ = 16;
constexpr int MS  = B_ * S_;      // 32832 rows, exact multiple of 16
constexpr int SP  = 544;          // key length padded to 34 tiles of 16

__device__ __forceinline__ v16bf pack_bf16x16(float4 a0, float4 a1, float4 a2, float4 a3) {
  v16bf r;
  r[0]=(__bf16)a0.x;  r[1]=(__bf16)a0.y;  r[2]=(__bf16)a0.z;  r[3]=(__bf16)a0.w;
  r[4]=(__bf16)a1.x;  r[5]=(__bf16)a1.y;  r[6]=(__bf16)a1.z;  r[7]=(__bf16)a1.w;
  r[8]=(__bf16)a2.x;  r[9]=(__bf16)a2.y;  r[10]=(__bf16)a2.z; r[11]=(__bf16)a2.w;
  r[12]=(__bf16)a3.x; r[13]=(__bf16)a3.y; r[14]=(__bf16)a3.z; r[15]=(__bf16)a3.w;
  return r;
}

// ---------------------------------------------------------------------------
// Embedding: feats(B,S,3) @ W_embed^T + b_embed  -> enc(B,S,D)
// ---------------------------------------------------------------------------
__global__ void embed_kernel(const float* __restrict__ loc, const float* __restrict__ start,
                             const float* __restrict__ weight, const float* __restrict__ We,
                             const float* __restrict__ be, float* __restrict__ enc) {
  int idx = blockIdx.x * blockDim.x + threadIdx.x;
  const int total = B_ * S_ * D_;
  if (idx >= total) return;
  int d   = idx % D_;
  int row = idx / D_;
  int s   = row % S_;
  int b   = row / S_;
  float f0, f1, f2;
  if (s == 0) {
    f0 = start[b * 2];  f1 = start[b * 2 + 1];  f2 = 0.f;
  } else {
    int n = s - 1;
    f0 = loc[((size_t)b * N_ + n) * 2];
    f1 = loc[((size_t)b * N_ + n) * 2 + 1];
    f2 = weight[(size_t)b * N_ + n];
  }
  enc[idx] = f0 * We[d * 3 + 0] + f1 * We[d * 3 + 1] + f2 * We[d * 3 + 2] + be[d];
}

// ---------------------------------------------------------------------------
// WMMA GEMM:  Y[M,N] = X[M,K] * W[N,K]^T + bias[N]  (optional ReLU)
// One block = one 16-row M strip. A strip staged in LDS (bf16) once, shared
// by all 8 waves; each wave sweeps N tiles. K templated (128/512) so the
// K-loop unrolls into dense back-to-back v_wmma.
// Operand packing per ISA 7.12.2 (16-bit 16x32 layout):
//   lane L: m/n = L&15, half = L>>4; element i <-> k = (i<8?0:16)+half*8+(i&7)
// ---------------------------------------------------------------------------
template <int K>
__global__ void __launch_bounds__(256)
gemm_bias_kernel(const float* __restrict__ X, const float* __restrict__ W,
                 const float* __restrict__ bias, float* __restrict__ Y,
                 int N, int relu) {
  __shared__ __bf16 Ash[16 * K];

  const int lane = threadIdx.x & 31;
  const int wave = threadIdx.x >> 5;
  const int half = lane >> 4;
  const int mr   = lane & 15;
  const int tm   = blockIdx.x;

  // cooperative A-strip load+convert: 16 rows x K floats -> bf16 in LDS
  {
    const float4* X4 = (const float4*)(X + (size_t)tm * 16 * K);
    const int nchunks = 16 * K / 4;
    for (int i = threadIdx.x; i < nchunks; i += 256) {
      float4 v = X4[i];
      __bf16* dst = &Ash[i * 4];
      dst[0] = (__bf16)v.x; dst[1] = (__bf16)v.y;
      dst[2] = (__bf16)v.z; dst[3] = (__bf16)v.w;
    }
  }
  __syncthreads();

  const int tilesN = N >> 4;
  for (int tn = wave; tn < tilesN; tn += 8) {
    const float* wr = W + (size_t)(tn * 16 + mr) * K;
    v8f acc = {};
#pragma unroll 4
    for (int k0 = 0; k0 < K; k0 += 32) {
      // A operand from LDS: two aligned 16B ds_load_b128 reads
      v8bf alo = *(const v8bf*)(&Ash[mr * K + k0 + half * 8]);
      v8bf ahi = *(const v8bf*)(&Ash[mr * K + k0 + 16 + half * 8]);
      v16bf a;
#pragma unroll
      for (int i = 0; i < 8; ++i) { a[i] = alo[i]; a[i + 8] = ahi[i]; }
      // B operand: weight row from global (L2 resident), fp32 -> bf16
      const float4* wa = (const float4*)(wr + k0 + half * 8);
      const float4* wb = (const float4*)(wr + k0 + 16 + half * 8);
      v16bf bm = pack_bf16x16(wa[0], wa[1], wb[0], wb[1]);
      acc = __builtin_amdgcn_wmma_f32_16x16x32_bf16(false, a, false, bm,
                                                    (short)0, acc, false, false);
    }
    const int n = tn * 16 + mr;
    const float bv = bias ? bias[n] : 0.f;
#pragma unroll
    for (int r = 0; r < 8; ++r) {
      float v = acc[r] + bv;
      if (relu) v = fmaxf(v, 0.f);
      Y[(size_t)(tm * 16 + r + 8 * half) * N + n] = v;   // C layout: row r+8*half, col lane&15
    }
  }
}

// ---------------------------------------------------------------------------
// Flash attention, one block per (b, head). dh=16 zero-padded to K=32.
// K/V staged in LDS as bf16 (SPx16 each). Per wave: 16-query tile, online
// softmax; P re-laid out C->A through per-wave LDS; PV via WMMA.
// ---------------------------------------------------------------------------
__global__ void __launch_bounds__(256)
attn_kernel(const float* __restrict__ qkv, float* __restrict__ out) {
  __shared__ __bf16 Ksh[SP * DH_];
  __shared__ __bf16 Vsh[SP * DH_];
  __shared__ __bf16 Psh[8][16 * 32];

  const int bh = blockIdx.x;
  const int b = bh / H_, h = bh % H_;
  const int lane = threadIdx.x & 31, wave = threadIdx.x >> 5;
  const int half = lane >> 4, mr = lane & 15;
  const int rowstride = 3 * D_;

  for (int s = threadIdx.x; s < SP; s += 256) {
    if (s < S_) {
      const float* kp = qkv + ((size_t)b * S_ + s) * rowstride + D_ + h * DH_;
      const float* vp = kp + D_;
      const float4* k4 = (const float4*)kp;
      const float4* v4 = (const float4*)vp;
#pragma unroll
      for (int j = 0; j < 4; ++j) {
        float4 kk = k4[j], vv = v4[j];
        Ksh[s * DH_ + j * 4 + 0] = (__bf16)kk.x;  Vsh[s * DH_ + j * 4 + 0] = (__bf16)vv.x;
        Ksh[s * DH_ + j * 4 + 1] = (__bf16)kk.y;  Vsh[s * DH_ + j * 4 + 1] = (__bf16)vv.y;
        Ksh[s * DH_ + j * 4 + 2] = (__bf16)kk.z;  Vsh[s * DH_ + j * 4 + 2] = (__bf16)vv.z;
        Ksh[s * DH_ + j * 4 + 3] = (__bf16)kk.w;  Vsh[s * DH_ + j * 4 + 3] = (__bf16)vv.w;
      }
    } else {
#pragma unroll
      for (int d = 0; d < DH_; ++d) { Ksh[s * DH_ + d] = (__bf16)0.f; Vsh[s * DH_ + d] = (__bf16)0.f; }
    }
  }
  __syncthreads();

  __bf16* P = &Psh[wave][0];
  for (int qt = wave; qt * 16 < S_; qt += 8) {
    int qrow = qt * 16 + mr;
    if (qrow >= S_) qrow = S_ - 1;                  // clamped rows computed, never stored
    const float* qp = qkv + ((size_t)b * S_ + qrow) * rowstride + h * DH_;
    v16bf aq;
#pragma unroll
    for (int i = 0; i < 8; ++i) {                   // k = half*8+i (<16 real), upper half = pad
      aq[i] = (__bf16)qp[half * 8 + i];
      aq[i + 8] = (__bf16)0.f;
    }
    float m_run[8], l_run[8];
    v8f acc = {};
#pragma unroll
    for (int r = 0; r < 8; ++r) { m_run[r] = -1e30f; l_run[r] = 0.f; }

    for (int kp2 = 0; kp2 < SP / 32; ++kp2) {
      const int kb = kp2 * 32;
      v8f s0 = {}, s1 = {};
      {
        v16bf bk;
#pragma unroll
        for (int i = 0; i < 8; ++i) { bk[i] = Ksh[(kb + mr) * DH_ + half * 8 + i]; bk[i + 8] = (__bf16)0.f; }
        s0 = __builtin_amdgcn_wmma_f32_16x16x32_bf16(false, aq, false, bk, (short)0, s0, false, false);
      }
      {
        v16bf bk;
#pragma unroll
        for (int i = 0; i < 8; ++i) { bk[i] = Ksh[(kb + 16 + mr) * DH_ + half * 8 + i]; bk[i + 8] = (__bf16)0.f; }
        s1 = __builtin_amdgcn_wmma_f32_16x16x32_bf16(false, aq, false, bk, (short)0, s1, false, false);
      }
      const float scale = 0.25f;                    // 1/sqrt(dh), dh=16
      const int key0 = kb + mr, key1 = kb + 16 + mr;
#pragma unroll
      for (int r = 0; r < 8; ++r) {
        float v0 = (key0 < S_) ? s0[r] * scale : -1e30f;
        float v1 = (key1 < S_) ? s1[r] * scale : -1e30f;
        float rowm = fmaxf(v0, v1);
#pragma unroll
        for (int o = 1; o < 16; o <<= 1) rowm = fmaxf(rowm, __shfl_xor(rowm, o, 32));
        float mnew = fmaxf(m_run[r], rowm);
        float p0 = __expf(v0 - mnew);
        float p1 = __expf(v1 - mnew);
        float rs = p0 + p1;
#pragma unroll
        for (int o = 1; o < 16; o <<= 1) rs += __shfl_xor(rs, o, 32);
        float fac = __expf(m_run[r] - mnew);
        l_run[r] = l_run[r] * fac + rs;
        acc[r] *= fac;
        m_run[r] = mnew;
        P[(r + 8 * half) * 32 + mr]      = (__bf16)p0;   // C layout -> dense 16x32 in LDS
        P[(r + 8 * half) * 32 + 16 + mr] = (__bf16)p1;
      }
      asm volatile("s_wait_dscnt 0" ::: "memory");       // wave-local LDS RAW fence
      v16bf ap, bv;
#pragma unroll
      for (int i = 0; i < 8; ++i) {                      // A layout read-back
        ap[i]     = P[mr * 32 + half * 8 + i];
        ap[i + 8] = P[mr * 32 + 16 + half * 8 + i];
      }
#pragma unroll
      for (int i = 0; i < 8; ++i) {                      // V as B operand (32 keys x 16 dh)
        bv[i]     = Vsh[(kb + half * 8 + i) * DH_ + mr];
        bv[i + 8] = Vsh[(kb + 16 + half * 8 + i) * DH_ + mr];
      }
      acc = __builtin_amdgcn_wmma_f32_16x16x32_bf16(false, ap, false, bv, (short)0, acc, false, false);
    }
#pragma unroll
    for (int r = 0; r < 8; ++r) {
      int q = qt * 16 + r + 8 * half;
      if (q < S_)
        out[((size_t)b * S_ + q) * D_ + h * DH_ + mr] = acc[r] / fmaxf(l_run[r], 1e-20f);
    }
  }
}

// ---------------------------------------------------------------------------
// x = LayerNorm(x + y) * g + b  — one 128-thread block per row
// ---------------------------------------------------------------------------
__global__ void __launch_bounds__(128)
add_ln_kernel(float* __restrict__ x, const float* __restrict__ y,
              const float* __restrict__ g, const float* __restrict__ bta) {
  const int row = blockIdx.x, t = threadIdx.x;
  const int lane = t & 31, wave = t >> 5;
  float v = x[(size_t)row * D_ + t] + y[(size_t)row * D_ + t];
  float s = v, q = v * v;
#pragma unroll
  for (int o = 16; o; o >>= 1) { s += __shfl_xor(s, o, 32); q += __shfl_xor(q, o, 32); }
  __shared__ float ps[4], pq[4];
  if (lane == 0) { ps[wave] = s; pq[wave] = q; }
  __syncthreads();
  s = ps[0] + ps[1] + ps[2] + ps[3];
  q = pq[0] + pq[1] + pq[2] + pq[3];
  float mean = s * (1.f / 128.f);
  float var  = q * (1.f / 128.f) - mean * mean;
  x[(size_t)row * D_ + t] = (v - mean) * rsqrtf(var + 1e-5f) * g[t] + bta[t];
}

// ---------------------------------------------------------------------------
// Greedy pointer decode: one block per batch element, 512 sequential steps.
// ---------------------------------------------------------------------------
__global__ void __launch_bounds__(256)
decode_kernel(const float* __restrict__ enc, const float* __restrict__ W1h,
              const float* __restrict__ W2, const float* __restrict__ vw,
              int* __restrict__ route, float* __restrict__ lps) {
  __shared__ float qsh[D_], dhsh[D_], vsh[D_];
  __shared__ float scores[S_];
  __shared__ unsigned char msk[S_];
  __shared__ float rval[256];
  __shared__ int   ridx[256];

  const int b = blockIdx.x, t = threadIdx.x;
  const float* encb = enc + (size_t)b * S_ * D_;
  const float* w1hb = W1h + (size_t)b * S_ * D_;

  if (t < D_) { vsh[t] = vw[t]; dhsh[t] = encb[t]; }
  for (int s = t; s < S_; s += 256) msk[s] = (s == 0);
  __syncthreads();

  for (int step = 0; step < N_; ++step) {
    // q = dec_h @ W2^T
    if (t < D_) {
      const float* w2r = W2 + (size_t)t * D_;
      float acc = 0.f;
#pragma unroll 4
      for (int k = 0; k < D_; ++k) acc += dhsh[k] * w2r[k];
      qsh[t] = acc;
    }
    __syncthreads();
    // scores[s] = v . tanh(W1h[s] + q), masked
    for (int s = t; s < S_; s += 256) {
      const float* wr = w1hb + (size_t)s * D_;
      if (s + 256 < S_) __builtin_prefetch(w1hb + (size_t)(s + 256) * D_, 0, 3);
      float acc = 0.f;
      for (int d = 0; d < D_; ++d) acc += vsh[d] * tanhf(wr[d] + qsh[d]);
      scores[s] = msk[s] ? -1e30f : acc;
    }
    __syncthreads();
    // argmax (first max wins)
    float bm = -1e31f; int bi = 0x7fffffff;
    for (int s = t; s < S_; s += 256) {
      float v = scores[s];
      if (v > bm || (v == bm && s < bi)) { bm = v; bi = s; }
    }
    rval[t] = bm; ridx[t] = bi;
    __syncthreads();
    for (int o = 128; o; o >>= 1) {
      if (t < o) {
        float v = rval[t + o]; int i2 = ridx[t + o];
        if (v > rval[t] || (v == rval[t] && i2 < ridx[t])) { rval[t] = v; ridx[t] = i2; }
      }
      __syncthreads();
    }
    const float mx = rval[0];
    const int  idx = ridx[0];
    __syncthreads();
    // log-softmax denominator
    float se = 0.f;
    for (int s = t; s < S_; s += 256) se += __expf(scores[s] - mx);
    rval[t] = se;
    __syncthreads();
    for (int o = 128; o; o >>= 1) { if (t < o) rval[t] += rval[t + o]; __syncthreads(); }
    if (t == 0) {
      route[(size_t)b * N_ + step] = idx;
      lps[(size_t)b * N_ + step]   = scores[idx] - mx - logf(rval[0]);
      msk[idx] = 1;
    }
    __syncthreads();
    if (t < D_) dhsh[t] = encb[(size_t)idx * D_ + t];
    __syncthreads();
  }
}

// ---------------------------------------------------------------------------
// Host launch. Input order = setup_inputs() insertion order:
//  0 loc, 1 start, 2 weight, 3 W_embed, 4 b_embed,
//  5..40: per layer l (0..2): w_in,b_in,w_out,b_out,w_ff1,b_ff1,w_ff2,b_ff2,
//                             ln1_g,ln1_b,ln2_g,ln2_b
//  41 W1, 42 W2, 43 v_w
// Output: B*N int route, then B*N float lps.
// ---------------------------------------------------------------------------
extern "C" void kernel_launch(void* const* d_in, const int* in_sizes, int n_in,
                              void* d_out, int out_size, void* d_ws, size_t ws_size,
                              hipStream_t stream) {
  (void)in_sizes; (void)n_in; (void)out_size; (void)ws_size;

  const float* loc    = (const float*)d_in[0];
  const float* start  = (const float*)d_in[1];
  const float* weight = (const float*)d_in[2];
  const float* We     = (const float*)d_in[3];
  const float* be     = (const float*)d_in[4];
  const float* W1     = (const float*)d_in[41];
  const float* W2     = (const float*)d_in[42];
  const float* vw     = (const float*)d_in[43];

  float* ws   = (float*)d_ws;
  size_t off  = 0;
  float* enc  = ws + off;  off += (size_t)MS * D_;
  float* qkv  = ws + off;  off += (size_t)MS * 3 * D_;
  float* attn = ws + off;  off += (size_t)MS * D_;
  float* ybuf = ws + off;  off += (size_t)MS * D_;
  float* hbuf = ws + off;  off += (size_t)MS * 4 * D_;
  float* w1h  = ws + off;

  {
    const int total = B_ * S_ * D_;
    embed_kernel<<<(total + 255) / 256, 256, 0, stream>>>(loc, start, weight, We, be, enc);
  }

  const int mStrips = MS / 16;   // 2052 blocks, one 16-row strip each
  auto gemm = [&](const float* X, const float* W, const float* bias, float* Y,
                  int N, int K, int relu) {
    if (K == 128)
      gemm_bias_kernel<128><<<mStrips, 256, 0, stream>>>(X, W, bias, Y, N, relu);
    else
      gemm_bias_kernel<512><<<mStrips, 256, 0, stream>>>(X, W, bias, Y, N, relu);
  };

  for (int l = 0; l < 3; ++l) {
    const float* w_in  = (const float*)d_in[5 + 12 * l + 0];
    const float* b_in  = (const float*)d_in[5 + 12 * l + 1];
    const float* w_out = (const float*)d_in[5 + 12 * l + 2];
    const float* b_out = (const float*)d_in[5 + 12 * l + 3];
    const float* w_ff1 = (const float*)d_in[5 + 12 * l + 4];
    const float* b_ff1 = (const float*)d_in[5 + 12 * l + 5];
    const float* w_ff2 = (const float*)d_in[5 + 12 * l + 6];
    const float* b_ff2 = (const float*)d_in[5 + 12 * l + 7];
    const float* ln1g  = (const float*)d_in[5 + 12 * l + 8];
    const float* ln1b  = (const float*)d_in[5 + 12 * l + 9];
    const float* ln2g  = (const float*)d_in[5 + 12 * l + 10];
    const float* ln2b  = (const float*)d_in[5 + 12 * l + 11];

    gemm(enc, w_in, b_in, qkv, 3 * D_, D_, 0);
    attn_kernel<<<B_ * H_, 256, 0, stream>>>(qkv, attn);
    gemm(attn, w_out, b_out, ybuf, D_, D_, 0);
    add_ln_kernel<<<MS, 128, 0, stream>>>(enc, ybuf, ln1g, ln1b);
    gemm(enc, w_ff1, b_ff1, hbuf, 4 * D_, D_, 1);
    gemm(hbuf, w_ff2, b_ff2, ybuf, D_, 4 * D_, 0);
    add_ln_kernel<<<MS, 128, 0, stream>>>(enc, ybuf, ln2g, ln2b);
  }

  gemm(enc, W1, nullptr, w1h, D_, D_, 0);

  int*   route = (int*)d_out;
  float* lps   = (float*)d_out + (size_t)B_ * N_;
  decode_kernel<<<B_, 256, 0, stream>>>(enc, w1h, W2, vw, route, lps);
}
